// Arch5GraphEncoder_8254927143545
// MI455X (gfx1250) — compile-verified, matrix operand order checked by reading, not run on hardware
//
#include <hip/hip_runtime.h>
#include <hip/hip_bf16.h>

typedef __attribute__((ext_vector_type(16))) _Float16 v16h;
typedef __attribute__((ext_vector_type(8)))  float    v8f;

#define NN   30000
#define KK   2
#define TT   (NN*KK)
#define NSN  (TT*4)
#define ESN  (NSN*3)
#define EE   (NN*16)
#define GG   300
#define HH   128
#define LL   4
#define BN_EPS_C 1e-5f

// ---------------------------------------------------------------------------
// Weight prep: f32 [128][128] (row-major, z@W -> W[k][n]) into per-lane WMMA
// B-fragment layout. Fragment (ntile, kstep): lane holds col n = n0+(lane&15),
// khalf = lane>>4; element pair v<4 -> k = k0+kh*8+2v+{0,1};
// v>=4 -> k = k0+16+kh*8+2(v-4)+{0,1}. Lane's 16 f16 stored contiguously.
// ---------------------------------------------------------------------------
__global__ void wprep_kernel(const float* __restrict__ W, _Float16* __restrict__ out) {
    int idx = blockIdx.x * blockDim.x + threadIdx.x;   // 16384 threads
    if (idx >= HH * HH) return;
    int e    = idx & 15;
    int lane = (idx >> 4) & 31;
    int ks   = (idx >> 9) & 3;
    int nt   = (idx >> 11) & 7;
    int n  = nt * 16 + (lane & 15);
    int kh = lane >> 4;
    int v  = e >> 1, p = e & 1;
    int k  = ks * 32 + kh * 8 + p + ((v < 4) ? (2 * v) : (16 + 2 * (v - 4)));
    out[idx] = (_Float16)W[k * HH + n];
}

// hs = atom_table[x[sub_node_map[node]]]
__global__ void gather_kernel(const float* __restrict__ atomtab, const int* __restrict__ x,
                              const int* __restrict__ snm, float* __restrict__ hs, int total) {
    int idx = blockIdx.x * blockDim.x + threadIdx.x;
    if (idx >= total) return;
    int node = idx >> 7, c = idx & 127;
    hs[idx] = atomtab[x[snm[node]] * HH + c];
}

// agg[dst] += h[src] over edges (edge-parallel, 32 lanes x 4 feats per edge)
__global__ void edgeagg_kernel(const float* __restrict__ h, const int* __restrict__ ei,
                               int nE, float* __restrict__ agg) {
    int idx = blockIdx.x * blockDim.x + threadIdx.x;
    if (idx >= nE * 32) return;
    int e = idx >> 5, lane = idx & 31;
    int src = ei[e], dst = ei[nE + e];
    int c0 = lane * 4;
    const float4 v = *(const float4*)(h + (size_t)src * HH + c0);
    float* a = agg + (size_t)dst * HH + c0;
    atomicAdd(a + 0, v.x); atomicAdd(a + 1, v.y);
    atomicAdd(a + 2, v.z); atomicAdd(a + 3, v.w);
}

// ---------------------------------------------------------------------------
// Fused GIN MLP: one wave32 per 16 nodes.
// z = (1+eps)*h + agg -> t = relu(z@W1+b1) -> y = t@W2+b2 (WMMA f16->f32)
// Also accumulates per-feature sum / sumsq for BatchNorm.
// ---------------------------------------------------------------------------
__global__ void __launch_bounds__(32)
gin_mlp_kernel(const float* __restrict__ hin, const float* __restrict__ agg,
               const _Float16* __restrict__ wf1, const _Float16* __restrict__ wf2,
               const float* __restrict__ b1, const float* __restrict__ b2,
               const float* __restrict__ epsp,
               float* __restrict__ yout, float* __restrict__ stats) {
    __shared__ float zbuf[16][HH];
    __shared__ float tbuf[16][HH];
    const int lane  = threadIdx.x;
    const int node0 = blockIdx.x * 16;
    const int kh    = lane >> 4;
    const int nl    = lane & 15;

    // Stage z tile into LDS
    const float epsv = 1.0f + epsp[0];
    const int c0 = lane * 4;
    for (int r = 0; r < 16; ++r) {
        const float4 hv = *(const float4*)(hin + (size_t)(node0 + r) * HH + c0);
        const float4 av = *(const float4*)(agg + (size_t)(node0 + r) * HH + c0);
        zbuf[r][c0 + 0] = epsv * hv.x + av.x;
        zbuf[r][c0 + 1] = epsv * hv.y + av.y;
        zbuf[r][c0 + 2] = epsv * hv.z + av.z;
        zbuf[r][c0 + 3] = epsv * hv.w + av.w;
    }
    __syncthreads();

    // A fragments for GEMM1 (16x32 f16 layout; row = lane&15, K-half by lane>=16)
    v16h afr[4];
#pragma unroll
    for (int ks = 0; ks < 4; ++ks) {
#pragma unroll
        for (int j = 0; j < 8; ++j) {
            afr[ks][j]     = (_Float16)zbuf[nl][ks * 32 + kh * 8 + j];
            afr[ks][8 + j] = (_Float16)zbuf[nl][ks * 32 + 16 + kh * 8 + j];
        }
    }

    // GEMM1: t = relu(z@W1 + b1), D-layout -> LDS row-major
#pragma unroll
    for (int nt = 0; nt < 8; ++nt) {
        v8f acc = {};
#pragma unroll
        for (int ks = 0; ks < 4; ++ks) {
            const v16h bf = *(const v16h*)(wf1 + ((size_t)(nt * 4 + ks) * 32 + lane) * 16);
            acc = __builtin_amdgcn_wmma_f32_16x16x32_f16(false, afr[ks], false, bf,
                                                         (short)0, acc, false, false);
        }
        const int col = nt * 16 + nl;
        const float bv = b1[col];
#pragma unroll
        for (int i = 0; i < 8; ++i)
            tbuf[i + kh * 8][col] = fmaxf(acc[i] + bv, 0.0f);
    }
    __syncthreads();

    // A fragments for GEMM2 from t
    v16h afr2[4];
#pragma unroll
    for (int ks = 0; ks < 4; ++ks) {
#pragma unroll
        for (int j = 0; j < 8; ++j) {
            afr2[ks][j]     = (_Float16)tbuf[nl][ks * 32 + kh * 8 + j];
            afr2[ks][8 + j] = (_Float16)tbuf[nl][ks * 32 + 16 + kh * 8 + j];
        }
    }

    // GEMM2: y = t@W2 + b2, write out + BN stats
#pragma unroll
    for (int nt = 0; nt < 8; ++nt) {
        v8f acc = {};
#pragma unroll
        for (int ks = 0; ks < 4; ++ks) {
            const v16h bf = *(const v16h*)(wf2 + ((size_t)(nt * 4 + ks) * 32 + lane) * 16);
            acc = __builtin_amdgcn_wmma_f32_16x16x32_f16(false, afr2[ks], false, bf,
                                                         (short)0, acc, false, false);
        }
        const int col = nt * 16 + nl;
        const float bv = b2[col];
        float s = 0.0f, ss = 0.0f;
#pragma unroll
        for (int i = 0; i < 8; ++i) {
            const float v = acc[i] + bv;
            yout[(size_t)(node0 + i + kh * 8) * HH + col] = v;
            s += v; ss += v * v;
        }
        atomicAdd(&stats[col], s);
        atomicAdd(&stats[HH + col], ss);
    }
}

// BatchNorm (training stats) + residual, in-place on hio
__global__ void bnres_kernel(float* __restrict__ hio, const float* __restrict__ y,
                             const float* __restrict__ stats, const float* __restrict__ gamma,
                             const float* __restrict__ beta, int rows) {
    int idx = blockIdx.x * blockDim.x + threadIdx.x;
    if (idx >= rows * HH) return;
    int c = idx & 127;
    float inv = 1.0f / (float)rows;
    float mu  = stats[c] * inv;
    float var = stats[HH + c] * inv - mu * mu;
    float rs  = rsqrtf(var + BN_EPS_C);
    hio[idx] = (y[idx] - mu) * rs * gamma[c] + beta[c] + hio[idx];
}

// K=2 softmax-weighted mean of subgraph root embeddings
__global__ void wmean_kernel(const float* __restrict__ hs, const float* __restrict__ logp,
                             const int* __restrict__ ridx, float* __restrict__ h, int total) {
    int idx = blockIdx.x * blockDim.x + threadIdx.x;
    if (idx >= total) return;
    int i = idx >> 7, c = idx & 127;
    float lp0 = logp[2 * i] * 2.0f;       // / TEMP(0.5)
    float lp1 = logp[2 * i + 1] * 2.0f;
    float m  = fmaxf(lp0, lp1);
    float e0 = __expf(lp0 - m), e1 = __expf(lp1 - m);
    float w0 = e0 / (e0 + e1);
    int r0 = ridx[2 * i], r1 = ridx[2 * i + 1];
    h[idx] = w0 * hs[(size_t)r0 * HH + c] + (1.0f - w0) * hs[(size_t)r1 * HH + c];
}

// Sum-pool per graph
__global__ void readout_kernel(const float* __restrict__ h, const int* __restrict__ batch,
                               float* __restrict__ out, int total) {
    int idx = blockIdx.x * blockDim.x + threadIdx.x;
    if (idx >= total) return;
    int i = idx >> 7, c = idx & 127;
    atomicAdd(&out[batch[i] * HH + c], h[idx]);
}

extern "C" void kernel_launch(void* const* d_in, const int* in_sizes, int n_in,
                              void* d_out, int out_size, void* d_ws, size_t ws_size,
                              hipStream_t stream) {
    const float* atomtab = (const float*)d_in[0];
    const float* subW1 = (const float*)d_in[2];
    const float* subb1 = (const float*)d_in[3];
    const float* subW2 = (const float*)d_in[4];
    const float* subb2 = (const float*)d_in[5];
    const float* subeps = (const float*)d_in[6];
    const float* subg  = (const float*)d_in[7];
    const float* subbt = (const float*)d_in[8];
    const float* gW1 = (const float*)d_in[9];
    const float* gb1 = (const float*)d_in[10];
    const float* gW2 = (const float*)d_in[11];
    const float* gb2 = (const float*)d_in[12];
    const float* geps = (const float*)d_in[13];
    const float* gg  = (const float*)d_in[14];
    const float* gbt = (const float*)d_in[15];
    const float* logp = (const float*)d_in[16];
    const int* x    = (const int*)d_in[17];
    const int* eidx = (const int*)d_in[18];
    const int* snm  = (const int*)d_in[20];
    const int* seidx = (const int*)d_in[21];
    const int* ridx = (const int*)d_in[22];
    const int* batch = (const int*)d_in[24];
    float* out = (float*)d_out;

    // Workspace layout
    char* ws = (char*)d_ws;
    float* hs   = (float*)(ws);                                    // NS*128 f32
    float* aggy = (float*)(ws + (size_t)NSN * HH * 4);             // NS*128 f32 (agg, then y)
    float* hbuf = (float*)(ws + (size_t)2 * NSN * HH * 4);         // N*128 f32
    _Float16* wfrag = (_Float16*)(ws + (size_t)2 * NSN * HH * 4 + (size_t)NN * HH * 4);
    float* stats = (float*)((char*)wfrag + (size_t)16 * HH * HH * 2); // 256 f32

    // 1. Weight fragments: slots 0..7 = sub (W1,W2 x 4 layers), 8..15 = glob
    for (int l = 0; l < LL; ++l) {
        wprep_kernel<<<64, 256, 0, stream>>>(subW1 + (size_t)l * HH * HH, wfrag + (size_t)(l * 2 + 0) * HH * HH);
        wprep_kernel<<<64, 256, 0, stream>>>(subW2 + (size_t)l * HH * HH, wfrag + (size_t)(l * 2 + 1) * HH * HH);
        wprep_kernel<<<64, 256, 0, stream>>>(gW1 + (size_t)l * HH * HH, wfrag + (size_t)(8 + l * 2 + 0) * HH * HH);
        wprep_kernel<<<64, 256, 0, stream>>>(gW2 + (size_t)l * HH * HH, wfrag + (size_t)(8 + l * 2 + 1) * HH * HH);
    }

    // 2. hs = atom_table[x[sub_node_map]]
    {
        int total = NSN * HH;
        gather_kernel<<<(total + 255) / 256, 256, 0, stream>>>(atomtab, x, snm, hs, total);
    }

    // 3. Subgraph GIN layers
    for (int l = 0; l < LL; ++l) {
        hipMemsetAsync(aggy, 0, (size_t)NSN * HH * 4, stream);
        hipMemsetAsync(stats, 0, 2 * HH * 4, stream);
        edgeagg_kernel<<<(ESN * 32 + 255) / 256, 256, 0, stream>>>(hs, seidx, ESN, aggy);
        gin_mlp_kernel<<<NSN / 16, 32, 0, stream>>>(hs, aggy,
            wfrag + (size_t)(l * 2 + 0) * HH * HH, wfrag + (size_t)(l * 2 + 1) * HH * HH,
            subb1 + l * HH, subb2 + l * HH, subeps + l, aggy, stats);
        bnres_kernel<<<(NSN * HH + 255) / 256, 256, 0, stream>>>(hs, aggy, stats,
            subg + l * HH, subbt + l * HH, NSN);
    }

    // 4. Softmax-weighted mean of roots -> h [N,128]
    {
        int total = NN * HH;
        wmean_kernel<<<(total + 255) / 256, 256, 0, stream>>>(hs, logp, ridx, hbuf, total);
    }

    // 5. Global GIN layers
    for (int l = 0; l < LL; ++l) {
        hipMemsetAsync(aggy, 0, (size_t)NN * HH * 4, stream);
        hipMemsetAsync(stats, 0, 2 * HH * 4, stream);
        edgeagg_kernel<<<(EE * 32 + 255) / 256, 256, 0, stream>>>(hbuf, eidx, EE, aggy);
        gin_mlp_kernel<<<NN / 16, 32, 0, stream>>>(hbuf, aggy,
            wfrag + (size_t)(8 + l * 2 + 0) * HH * HH, wfrag + (size_t)(8 + l * 2 + 1) * HH * HH,
            gb1 + l * HH, gb2 + l * HH, geps + l, aggy, stats);
        bnres_kernel<<<(NN * HH + 255) / 256, 256, 0, stream>>>(hbuf, aggy, stats,
            gg + l * HH, gbt + l * HH, NN);
    }

    // 6. Readout
    hipMemsetAsync(out, 0, (size_t)GG * HH * 4, stream);
    readout_kernel<<<(NN * HH + 255) / 256, 256, 0, stream>>>(hbuf, batch, out, NN * HH);
}
